// STAMP_77996606095968
// MI455X (gfx1250) — compile-verified, matrix operand order checked by reading, not run on hardware
//
#include <hip/hip_runtime.h>
#include <hip/hip_bf16.h>
#include <math.h>

typedef __attribute__((ext_vector_type(16))) _Float16 v16h;
typedef __attribute__((ext_vector_type(4)))  _Float16 v4h;
typedef __attribute__((ext_vector_type(8)))  float    v8f;
typedef __attribute__((ext_vector_type(4)))  float    v4f;

#define BB 4096
#define LL 200
#define LP 208   // padded to 13 M-tiles of 16
#define DD 128

// LDS layout (bytes)
#define OFF_XI32 0                         // float [200*128] = 102400
#define OFF_XI16 102400                    // _Float16[208*128] = 53248
#define OFF_ALP  (OFF_XI16 + 53248)        // float [208] = 832
#define OFF_XT   (OFF_ALP + 832)           // float [128]
#define OFF_MS   (OFF_XT + 512)            // float [128]
#define OFF_C    (OFF_MS + 512)            // float [128]
#define OFF_MA   (OFF_C + 512)             // float [128]
#define OFF_RED  (OFF_MA + 512)            // float [256] partials
#define OFF_RED2 (OFF_RED + 1024)          // float [256] partials
#define OFF_SEQ  (OFF_RED2 + 1024)         // int  [208]
#define SMEM_BYTES (OFF_SEQ + 832)         // = 161408 (2 blocks/WGP at 2KB granules)

// K index inside a 16x32 f16 A/B fragment: element i of v16h, lane-half g.
// ISA 7.12.2: lane-half 0 holds K {0..7, 16..23}, lane-half 1 holds {8..15, 24..31}.
__device__ __forceinline__ int kmap(int i, int g) {
    int v = i >> 1, h = i & 1;
    int base = (v < 4) ? (2 * v) : (16 + 2 * (v - 4));
    return base + h + 8 * g;
}

// x + dpp_permuted(x): VALU-only lane exchange, no LDS traffic.
template <int CTRL>
__device__ __forceinline__ float dpp_add(float x) {
    int y = __builtin_amdgcn_update_dpp(0, __float_as_int(x), CTRL, 0xF, 0xF, true);
    return x + __int_as_float(y);
}
// Sum across a 16-lane half: xor1, xor2 (quad_perm), then half/row mirror
// (equivalent to xor4 / xor8 once quads and octs are uniform).
__device__ __forceinline__ float row16_sum(float x) {
    x = dpp_add<0xB1>(x);    // quad_perm [1,0,3,2]  == xor 1
    x = dpp_add<0x4E>(x);    // quad_perm [2,3,0,1]  == xor 2
    x = dpp_add<0x141>(x);   // row_half_mirror      == xor 4 (post-uniform quads)
    x = dpp_add<0x140>(x);   // row_mirror           == xor 8 (post-uniform octs)
    return x;
}

__global__ void STAMP_narm_kernel(const int* __restrict__ seqs,
                                  const int* __restrict__ lengths,
                                  const float* __restrict__ emb,
                                  const float* __restrict__ W1,
                                  const float* __restrict__ W2,
                                  const float* __restrict__ W3,
                                  const float* __restrict__ b_a,
                                  const float* __restrict__ Wa,
                                  const float* __restrict__ Wb,
                                  float* __restrict__ out_xi,
                                  float* __restrict__ out_pred) {
    extern __shared__ __align__(16) char smem[];
    float*    s_xi32 = (float*)(smem + OFF_XI32);
    _Float16* s_xi16 = (_Float16*)(smem + OFF_XI16);
    float*    s_alp  = (float*)(smem + OFF_ALP);
    float*    s_xt   = (float*)(smem + OFF_XT);
    float*    s_ms   = (float*)(smem + OFF_MS);
    float*    s_c    = (float*)(smem + OFF_C);
    float*    s_ma   = (float*)(smem + OFF_MA);
    float*    s_red  = (float*)(smem + OFF_RED);
    float*    s_red2 = (float*)(smem + OFF_RED2);
    int*      s_seq  = (int*)(smem + OFF_SEQ);

    const int b   = blockIdx.x;
    const int tid = threadIdx.x;

    // ---- Phase 0: session token ids + init alp ----
    for (int l = tid; l < LP; l += 256) {
        s_seq[l] = (l < LL) ? seqs[(size_t)b * LL + l] : 0;
        s_alp[l] = 0.0f;
    }
    __syncthreads();
    const int len = lengths[b];

    // ---- Phase 1: vectorized embedding gather -> LDS (f32 + f16), stream xi ----
    {
        const v4f* emb4   = (const v4f*)emb;
        v4f*       xi32_4 = (v4f*)s_xi32;
        v4h*       xi16_4 = (v4h*)s_xi16;
        v4f*       out4   = (v4f*)(out_xi + (size_t)b * (LL * DD));
        for (int idx = tid; idx < (LL * DD) / 4; idx += 256) {
            int l = idx >> 5, d4 = idx & 31;          // 32 float4 per row
            v4f v = emb4[(size_t)s_seq[l] * 32 + d4];
            xi32_4[idx] = v;
            v4h hv;
#pragma unroll
            for (int i = 0; i < 4; ++i) hv[i] = (_Float16)v[i];
            xi16_4[idx] = hv;
            out4[idx] = v;
        }
        v4h z;
#pragma unroll
        for (int i = 0; i < 4; ++i) z[i] = (_Float16)0.0f;
        for (int idx = (LL * DD) / 4 + tid; idx < (LP * DD) / 4; idx += 256)
            xi16_4[idx] = z;                           // zero-pad rows 200..207
    }
    __syncthreads();

    // ---- Phase 2: mean interest ms and last item xt (256-thread split) ----
    {
        int d = tid & 127, h = tid >> 7;
        float acc = 0.0f;
        for (int l = h; l < LL; l += 2) acc += s_xi32[l * DD + d];
        s_red[tid] = acc;
    }
    __syncthreads();
    if (tid < DD) {
        s_ms[tid] = (s_red[tid] + s_red[tid + 128]) / (float)len;
        s_xt[tid] = s_xi32[(len - 1) * DD + tid];
    }
    __syncthreads();

    // ---- Phase 3: c[j] = (xt @ W2^T + ms @ W3^T)[j], split + vectorized ----
    {
        int j = tid & 127, h = tid >> 7;
        const v4f* w2r = (const v4f*)(W2 + (size_t)j * DD + h * 64);
        const v4f* w3r = (const v4f*)(W3 + (size_t)j * DD + h * 64);
        const v4f* xt4 = (const v4f*)(s_xt + h * 64);
        const v4f* ms4 = (const v4f*)(s_ms + h * 64);
        float acc = 0.0f;
#pragma unroll 4
        for (int q = 0; q < 16; ++q) {
            v4f a = w2r[q], c = xt4[q], d = w3r[q], e = ms4[q];
#pragma unroll
            for (int i = 0; i < 4; ++i) acc += c[i] * a[i] + e[i] * d[i];
        }
        s_red[tid] = acc;
    }
    __syncthreads();
    if (tid < DD) s_c[tid] = s_red[tid] + s_red[tid + 128];
    __syncthreads();

    // ---- Phase 4: WMMA GEMM res = xi @ W1^T (f16 in, f32 acc), then
    //      alp[l] += sum_n sigmoid(res[l,n]+c[n]) * b_a[n] ----
    const int lane = tid & 31;
    const int wv   = tid >> 5;     // 8 waves, one N-tile each
    const int col  = lane & 15;    // N within tile / M row within A
    const int g    = lane >> 4;    // lane-half -> K group
    const int n    = wv * 16 + col;

    const float ba = b_a[n];
    const float cj = s_c[n];

    // B fragments: column n of W1^T == row n of W1; resident for all M-tiles
    v16h bf[4];
#pragma unroll
    for (int kt = 0; kt < 4; ++kt) {
#pragma unroll
        for (int i = 0; i < 16; ++i)
            bf[kt][i] = (_Float16)W1[(size_t)n * DD + kt * 32 + kmap(i, g)];
    }

    for (int mt = 0; mt < LP / 16; ++mt) {
        const int m = mt * 16 + col;
        v16h a[4];
#pragma unroll
        for (int kt = 0; kt < 4; ++kt) {
#pragma unroll
            for (int i = 0; i < 16; ++i)
                a[kt][i] = s_xi16[m * DD + kt * 32 + kmap(i, g)];
        }
        v8f acc = {};
#pragma unroll
        for (int kt = 0; kt < 4; ++kt)
            acc = __builtin_amdgcn_wmma_f32_16x16x32_f16(
                false, a[kt], false, bf[kt], (short)0, acc, false, false);

        // Batched: 8 independent sigmoid chains, 8 independent DPP reductions,
        // one exec-masked block of 8 ds_add_f32.
        float p[8];
#pragma unroll
        for (int r = 0; r < 8; ++r)
            p[r] = ba * __builtin_amdgcn_rcpf(1.0f + __expf(-(acc[r] + cj)));
#pragma unroll
        for (int r = 0; r < 8; ++r)
            p[r] = row16_sum(p[r]);
        if (col == 0) {
#pragma unroll
            for (int r = 0; r < 8; ++r)
                atomicAdd(&s_alp[mt * 16 + r + 8 * g], p[r]);
        }
    }
    __syncthreads();

    // ---- Phase 5: mask alp where token == padding ----
    for (int l = tid; l < LL; l += 256)
        if (s_seq[l] == 0) s_alp[l] = 0.0f;
    __syncthreads();

    // ---- Phase 6: ma = alp^T xi + ms (256-thread split) ----
    {
        int d = tid & 127, h = tid >> 7;
        float acc = 0.0f;
        for (int l = h; l < LL; l += 2) acc += s_alp[l] * s_xi32[l * DD + d];
        s_red[tid] = acc;
    }
    __syncthreads();
    if (tid < DD)
        s_ma[tid] = s_red[tid] + s_red[tid + 128] + s_ms[tid];
    __syncthreads();

    // ---- Phase 7: predictions = tanh(ma @ Wa^T) * tanh(xt @ Wb^T), split ----
    {
        int j = tid & 127, h = tid >> 7;
        const v4f* war = (const v4f*)(Wa + (size_t)j * DD + h * 64);
        const v4f* wbr = (const v4f*)(Wb + (size_t)j * DD + h * 64);
        const v4f* ma4 = (const v4f*)(s_ma + h * 64);
        const v4f* xt4 = (const v4f*)(s_xt + h * 64);
        float h1 = 0.0f, h2 = 0.0f;
#pragma unroll 4
        for (int q = 0; q < 16; ++q) {
            v4f a = war[q], c = ma4[q], d = wbr[q], e = xt4[q];
#pragma unroll
            for (int i = 0; i < 4; ++i) { h1 += c[i] * a[i]; h2 += e[i] * d[i]; }
        }
        s_red[tid]  = h1;
        s_red2[tid] = h2;
    }
    __syncthreads();
    if (tid < DD) {
        float h1 = s_red[tid] + s_red[tid + 128];
        float h2 = s_red2[tid] + s_red2[tid + 128];
        out_pred[(size_t)b * DD + tid] = tanhf(h1) * tanhf(h2);
    }
}

extern "C" void kernel_launch(void* const* d_in, const int* in_sizes, int n_in,
                              void* d_out, int out_size, void* d_ws, size_t ws_size,
                              hipStream_t stream) {
    (void)in_sizes; (void)n_in; (void)d_ws; (void)ws_size; (void)out_size;
    const int*   seqs    = (const int*)d_in[0];
    const int*   lengths = (const int*)d_in[1];
    const float* emb     = (const float*)d_in[2];
    const float* W1      = (const float*)d_in[3];
    const float* W2      = (const float*)d_in[4];
    const float* W3      = (const float*)d_in[5];
    const float* b_a     = (const float*)d_in[6];
    const float* Wa      = (const float*)d_in[7];
    const float* Wb      = (const float*)d_in[8];

    float* out_xi   = (float*)d_out;
    float* out_pred = out_xi + (size_t)BB * LL * DD;

    STAMP_narm_kernel<<<BB, 256, SMEM_BYTES, stream>>>(
        seqs, lengths, emb, W1, W2, W3, b_a, Wa, Wb, out_xi, out_pred);
}
